// Transformer_35227321762183
// MI455X (gfx1250) — matrix operand order repaired and checked by reading, not compiled, not run on hardware
//
#include <hip/hip_runtime.h>
#include <math.h>

#define B_    4
#define T_    512
#define S_    1536
#define HD    512
#define NH_   8
#define DH_   64
#define NBLK_ 6
#define DFF_  2048
#define MROWS (B_*S_)   // 6144

typedef __bf16 bhalf;
typedef __attribute__((ext_vector_type(16))) __bf16 v16bf;
typedef __attribute__((ext_vector_type(8)))  __bf16 v8bf;
typedef __attribute__((ext_vector_type(8)))  float  v8f;

static __device__ __forceinline__ v16bf cat16(v8bf lo, v8bf hi) {
  return __builtin_shufflevector(lo, hi, 0,1,2,3,4,5,6,7,8,9,10,11,12,13,14,15);
}
static __device__ __forceinline__ v8f wmma_bf16(v16bf a, v16bf b, v8f c) {
  // D = A(16x32 bf16) x B(32x16 bf16) + C(16x16 f32)
  return __builtin_amdgcn_wmma_f32_16x16x32_bf16(false, a, false, b, (short)0, c, false, false);
}
// Async copy global->LDS (16B per lane), tracked by ASYNCcnt.
// gfx1250 flat->LDS mapping: LDS byte offset = addr[31:0], so truncating the
// generic shared pointer gives the VDST LDS address.
static __device__ __forceinline__ void async_cp16(const bhalf* g, const __bf16* l) {
  unsigned la = (unsigned)(uintptr_t)l;
  asm volatile("global_load_async_to_lds_b128 %0, %1, off"
               :: "v"(la), "v"(g) : "memory");
}

// ---------------------------------------------------------------------------
// Weight convert + transpose: W[K][N] f32 -> Wt[N][K] bf16
// ---------------------------------------------------------------------------
__global__ void wcvt_t_kernel(const float* __restrict__ W, bhalf* __restrict__ Wt,
                              int K, int N) {
  int idx = blockIdx.x * 256 + threadIdx.x;
  int k = idx / N, n = idx % N;
  Wt[(size_t)n * K + k] = (bhalf)W[idx];
}

// ---------------------------------------------------------------------------
// Embedding + input LayerNorm. One block per (b, s) row of 512.
// ---------------------------------------------------------------------------
__global__ __launch_bounds__(256)
void embed_ln_kernel(const int* __restrict__ timesteps,
                     const float* __restrict__ s0g, const float* __restrict__ s1g,
                     const float* __restrict__ act, const float* __restrict__ te,
                     const float* __restrict__ Ws, const float* __restrict__ bs,
                     const float* __restrict__ Wa, const float* __restrict__ ba,
                     const float* __restrict__ eg, const float* __restrict__ eb,
                     float* __restrict__ h, bhalf* __restrict__ hbf) {
  const int row = blockIdx.x;              // 0..MROWS-1
  const int b = row / S_, s = row % S_;
  const int t = s / 3, st = s % 3;
  const int tid = threadIdx.x;

  const float* in; const float* W; const float* bias; int K;
  if (st == 0)      { in = s0g + (size_t)(b*T_ + t)*64; W = Ws; bias = bs; K = 64; }
  else if (st == 1) { in = s1g + (size_t)(b*T_ + t)*64; W = Ws; bias = bs; K = 64; }
  else              { in = act + (size_t)(b*T_ + t)*32; W = Wa; bias = ba; K = 32; }
  const float* terow = te + (size_t)timesteps[b*T_ + t] * HD;

  __shared__ float xin[64];
  __shared__ float rs[256], rq[256];
  if (tid < K) xin[tid] = in[tid];
  __syncthreads();

  float y[2];
  #pragma unroll
  for (int i = 0; i < 2; i++) {
    int n = tid + i*256;
    float a = bias[n] + terow[n];
    for (int kk = 0; kk < K; kk++) a += xin[kk] * W[kk*HD + n];
    y[i] = a;
  }
  rs[tid] = y[0] + y[1];
  rq[tid] = y[0]*y[0] + y[1]*y[1];
  __syncthreads();
  for (int stp = 128; stp > 0; stp >>= 1) {
    if (tid < stp) { rs[tid] += rs[tid+stp]; rq[tid] += rq[tid+stp]; }
    __syncthreads();
  }
  float mean = rs[0] * (1.f/512.f);
  float var  = rq[0] * (1.f/512.f) - mean*mean;
  float inv  = rsqrtf(var + 1e-5f);
  #pragma unroll
  for (int i = 0; i < 2; i++) {
    int n = tid + i*256;
    float o = (y[i] - mean) * inv * eg[n] + eb[n];
    size_t oi = (size_t)row*HD + n;
    h[oi] = o;
    hbf[oi] = (bhalf)o;
  }
}

// ---------------------------------------------------------------------------
// LayerNorm over 512: f32 in -> f32 out + bf16 out
// ---------------------------------------------------------------------------
__global__ __launch_bounds__(256)
void ln_kernel(const float* __restrict__ in, const float* __restrict__ g,
               const float* __restrict__ be, float* __restrict__ outf,
               bhalf* __restrict__ outb) {
  const int row = blockIdx.x;
  const int tid = threadIdx.x;
  const float* x = in + (size_t)row*HD;
  float v0 = x[tid], v1 = x[tid + 256];
  __shared__ float rs[256], rq[256];
  rs[tid] = v0 + v1;
  rq[tid] = v0*v0 + v1*v1;
  __syncthreads();
  for (int stp = 128; stp > 0; stp >>= 1) {
    if (tid < stp) { rs[tid] += rs[tid+stp]; rq[tid] += rq[tid+stp]; }
    __syncthreads();
  }
  float mean = rs[0] * (1.f/512.f);
  float var  = rq[0] * (1.f/512.f) - mean*mean;
  float inv  = rsqrtf(var + 1e-5f);
  float y[2] = {v0, v1};
  #pragma unroll
  for (int i = 0; i < 2; i++) {
    int n = tid + i*256;
    float o = (y[i] - mean) * inv * g[n] + be[n];
    size_t oi = (size_t)row*HD + n;
    outf[oi] = o;
    outb[oi] = (bhalf)o;
  }
}

// ---------------------------------------------------------------------------
// bf16 WMMA GEMM: C[M][N] = A[M][K] x Bt[N][K]^T (+bias, epilogue)
// Block tile 128x128, 8 waves (4x2), wave tile 32x64.
// K staged 64-wide through double-buffered LDS fed by async global->LDS
// copies (ASYNCcnt), so global fetch overlaps WMMA issue.
// EPI: 0 = bf16 store, 1 = gelu -> bf16 store, 2 = +res -> f32 store
// ---------------------------------------------------------------------------
template <int EPI>
__global__ __launch_bounds__(256)
void gemm_bf16_kernel(const bhalf* __restrict__ A, const bhalf* __restrict__ Bt,
                      const float* __restrict__ bias, void* __restrict__ outp,
                      const float* __restrict__ res, int M, int N, int K) {
  __shared__ __bf16 As[2][128][72];
  __shared__ __bf16 Bs[2][128][72];
  const int tid  = threadIdx.x;
  const int lane = tid & 31;
  const int wave = tid >> 5;
  const int wm = wave >> 1;      // 0..3 (M dir)
  const int wn = wave & 1;       // 0..1 (N dir)
  const int l15  = lane & 15;
  const int half = lane >> 4;
  const int rowBase = blockIdx.y * 128;
  const int colBase = blockIdx.x * 128;

  const int r_ld  = tid >> 3;            // this thread's LDS row (per chunk step 32)
  const int c8_ld = (tid & 7) << 3;      // this thread's 8-elem column chunk

  const v8f vzero = {0.f,0.f,0.f,0.f,0.f,0.f,0.f,0.f};
  v8f acc[2][4];
  #pragma unroll
  for (int i = 0; i < 2; i++)
    #pragma unroll
    for (int j = 0; j < 4; j++) acc[i][j] = vzero;

  // Issue one K-stage of async copies (8 async instructions per wave).
  auto issue_stage = [&](int buf, int k0) {
    #pragma unroll
    for (int i = 0; i < 4; i++) {
      int r = r_ld + i*32;
      async_cp16(A  + (size_t)(rowBase + r)*K + k0 + c8_ld, &As[buf][r][c8_ld]);
      async_cp16(Bt + (size_t)(colBase + r)*K + k0 + c8_ld, &Bs[buf][r][c8_ld]);
    }
  };

  const int nstage = K >> 6;
  issue_stage(0, 0);
  for (int s = 0; s < nstage; s++) {
    const int buf = s & 1;
    if (s + 1 < nstage) {
      issue_stage(buf ^ 1, (s + 1) << 6);
      // current stage's 8 copies done; next stage's 8 still in flight
      asm volatile("s_wait_asynccnt 0x8" ::: "memory");
    } else {
      asm volatile("s_wait_asynccnt 0x0" ::: "memory");
    }
    __syncthreads();
    #pragma unroll
    for (int kk = 0; kk < 64; kk += 32) {
      v16bf afr[2];
      #pragma unroll
      for (int mt = 0; mt < 2; mt++) {
        int r  = wm*32 + mt*16 + l15;
        int ks = kk + half*16;
        afr[mt] = cat16(*(const v8bf*)&As[buf][r][ks], *(const v8bf*)&As[buf][r][ks + 8]);
      }
      #pragma unroll
      for (int nt = 0; nt < 4; nt++) {
        int nr = wn*64 + nt*16 + l15;
        int ks = kk + half*16;
        v16bf bfr = cat16(*(const v8bf*)&Bs[buf][nr][ks], *(const v8bf*)&Bs[buf][nr][ks + 8]);
        acc[0][nt] = wmma_bf16(afr[0], bfr, acc[0][nt]);
        acc[1][nt] = wmma_bf16(afr[1], bfr, acc[1][nt]);
      }
    }
    __syncthreads();
  }

  #pragma unroll
  for (int nt = 0; nt < 4; nt++) {
    int col = colBase + wn*64 + nt*16 + l15;
    float bv = bias[col];
    #pragma unroll
    for (int mt = 0; mt < 2; mt++) {
      #pragma unroll
      for (int v = 0; v < 8; v++) {
        int row = rowBase + wm*32 + mt*16 + half*8 + v;
        float x = acc[mt][nt][v] + bv;
        size_t oi = (size_t)row * N + col;
        if (EPI == 0) {
          ((bhalf*)outp)[oi] = (bhalf)x;
        } else if (EPI == 1) {
          float ge = 0.5f * x * (1.f + erff(x * 0.70710678118f));
          ((bhalf*)outp)[oi] = (bhalf)ge;
        } else {
          ((float*)outp)[oi] = x + res[oi];
        }
      }
    }
  }
}

// ---------------------------------------------------------------------------
// V transpose: v[b][s][h*64+d] bf16 -> vt[b][h][d][s] bf16
// ---------------------------------------------------------------------------
__global__ void vtrans_kernel(const bhalf* __restrict__ v, bhalf* __restrict__ vt) {
  int idx = blockIdx.x * 256 + threadIdx.x;       // over MROWS*HD
  int c = idx & (HD - 1);
  int r = idx >> 9;                               // b*S_+s
  int b = r / S_, s = r % S_;
  int hh = c >> 6, d = c & 63;
  vt[(((size_t)(b*NH_ + hh)*DH_) + d) * S_ + s] = v[idx];
}

// ---------------------------------------------------------------------------
// Flash attention. One wave per (b, head, 16-query tile); 4 waves / block.
// ---------------------------------------------------------------------------
__global__ __launch_bounds__(128)
void attn_kernel(const bhalf* __restrict__ q, const bhalf* __restrict__ kM,
                 const bhalf* __restrict__ vt, bhalf* __restrict__ ctx) {
  __shared__ __bf16 P[4][16][40];                 // per-wave P staging (16x32)
  const int lane = threadIdx.x & 31;
  const int wave = threadIdx.x >> 5;
  const int l15  = lane & 15;
  const int half = lane >> 4;
  int gw = blockIdx.x * 4 + wave;
  const int qt = gw % (S_/16);  gw /= (S_/16);
  const int hh = gw % NH_;
  const int bb = gw / NH_;
  const int q0 = qt * 16;

  // Q fragments held in registers across the whole key loop
  const bhalf* qrow = q + ((size_t)(bb*S_ + q0 + l15))*HD + hh*DH_;
  v16bf qa[2];
  #pragma unroll
  for (int kk = 0; kk < 2; kk++) {
    int d0 = kk*32 + half*16;
    qa[kk] = cat16(*(const v8bf*)(qrow + d0), *(const v8bf*)(qrow + d0 + 8));
  }

  const v8f vzero = {0.f,0.f,0.f,0.f,0.f,0.f,0.f,0.f};
  v8f o[4];
  #pragma unroll
  for (int i = 0; i < 4; i++) o[i] = vzero;
  float mrun[8], lrun[8];
  #pragma unroll
  for (int v = 0; v < 8; v++) { mrun[v] = -1e30f; lrun[v] = 0.f; }

  const bhalf* kbase  = kM + (size_t)(bb*S_)*HD + hh*DH_;
  const bhalf* vtbase = vt + (size_t)(bb*NH_ + hh)*DH_*S_;

  for (int j0 = 0; j0 < q0 + 16; j0 += 32) {
    v8f s0 = vzero, s1 = vzero;
    #pragma unroll
    for (int kk = 0; kk < 2; kk++) {
      int d0 = kk*32 + half*16;
      const bhalf* kr0 = kbase + (size_t)(j0 + l15)*HD + d0;
      const bhalf* kr1 = kbase + (size_t)(j0 + 16 + l15)*HD + d0;
      v16bf kb0 = cat16(*(const v8bf*)kr0, *(const v8bf*)(kr0 + 8));
      v16bf kb1 = cat16(*(const v8bf*)kr1, *(const v8bf*)(kr1 + 8));
      s0 = wmma_bf16(qa[kk], kb0, s0);
      s1 = wmma_bf16(qa[kk], kb1, s1);
    }
    // online softmax update (rows live on 16-lane halves; xor masks<=8 stay inside)
    #pragma unroll
    for (int v = 0; v < 8; v++) {
      const int qi = q0 + half*8 + v;
      float x0 = s0[v] * 0.125f;
      float x1 = s1[v] * 0.125f;
      if (j0 + l15 > qi)      x0 = -1e30f;
      if (j0 + 16 + l15 > qi) x1 = -1e30f;
      float rmax = fmaxf(x0, x1);
      #pragma unroll
      for (int msk = 1; msk < 16; msk <<= 1) rmax = fmaxf(rmax, __shfl_xor(rmax, msk));
      float mnew  = fmaxf(mrun[v], rmax);
      float scale = __expf(mrun[v] - mnew);
      float p0 = __expf(x0 - mnew);
      float p1 = __expf(x1 - mnew);
      float rsum = p0 + p1;
      #pragma unroll
      for (int msk = 1; msk < 16; msk <<= 1) rsum += __shfl_xor(rsum, msk);
      lrun[v] = lrun[v]*scale + rsum;
      mrun[v] = mnew;
      #pragma unroll
      for (int nt = 0; nt < 4; nt++) o[nt][v] *= scale;
      int r = half*8 + v;
      P[wave][r][l15]      = (bhalf)p0;
      P[wave][r][16 + l15] = (bhalf)p1;
    }
    asm volatile("s_wait_dscnt 0" ::: "memory");   // cross-lane LDS transpose visible
    v16bf pa = cat16(*(const v8bf*)&P[wave][l15][half*16],
                     *(const v8bf*)&P[wave][l15][half*16 + 8]);
    #pragma unroll
    for (int nt = 0; nt < 4; nt++) {
      const bhalf* vr = vtbase + (size_t)(nt*16 + l15)*S_ + j0 + half*16;
      v16bf vb = cat16(*(const v8bf*)vr, *(const v8bf*)(vr + 8));
      o[nt] = wmma_bf16(pa, vb, o[nt]);
    }
  }
  #pragma unroll
  for (int nt = 0; nt < 4; nt++) {
    #pragma unroll
    for (int v = 0; v < 8; v++) {
      int r = half*8 + v;
      float val = o[nt][v] / lrun[v];
      ctx[((size_t)(bb*S_ + q0 + r))*HD + hh*DH_ + nt*16 + l15] = (bhalf)val;
    }
  }
}

// ---------------------------------------------------------------------------
// Final projection: take stream-1 rows, 512 -> 32
// ---------------------------------------------------------------------------
__global__ __launch_bounds__(32)
void fproj_kernel(const float* __restrict__ h, const float* __restrict__ Wp,
                  const float* __restrict__ bp, float* __restrict__ out) {
  int bt = blockIdx.x;                 // 0..B*T-1
  int n  = threadIdx.x;                // 0..31
  int b = bt / T_, t = bt % T_;
  const float* row = h + (size_t)(b*S_ + 3*t + 1) * HD;
  float acc = bp[n];
  for (int kk = 0; kk < HD; kk++) acc += row[kk] * Wp[kk*32 + n];
  out[(size_t)bt*32 + n] = acc;
}

// ---------------------------------------------------------------------------
extern "C" void kernel_launch(void* const* d_in, const int* in_sizes, int n_in,
                              void* d_out, int out_size, void* d_ws, size_t ws_size,
                              hipStream_t stream) {
  const int*   timesteps = (const int*)d_in[0];
  const float* state0 = (const float*)d_in[1];
  const float* state1 = (const float*)d_in[2];
  const float* actions= (const float*)d_in[3];
  const float* time_emb=(const float*)d_in[4];
  const float* Ws = (const float*)d_in[5];
  const float* bs = (const float*)d_in[6];
  const float* Wa = (const float*)d_in[7];
  const float* ba = (const float*)d_in[8];
  const float* Wq = (const float*)d_in[9];
  const float* bq = (const float*)d_in[10];
  const float* Wk = (const float*)d_in[11];
  const float* bk = (const float*)d_in[12];
  const float* Wv = (const float*)d_in[13];
  const float* bv = (const float*)d_in[14];
  const float* Wo = (const float*)d_in[15];
  const float* bo = (const float*)d_in[16];
  const float* W1 = (const float*)d_in[17];
  const float* b1 = (const float*)d_in[18];
  const float* W2 = (const float*)d_in[19];
  const float* b2 = (const float*)d_in[20];
  const float* ln1g = (const float*)d_in[21];
  const float* ln1b = (const float*)d_in[22];
  const float* ln2g = (const float*)d_in[23];
  const float* ln2b = (const float*)d_in[24];
  const float* elng = (const float*)d_in[25];
  const float* elnb = (const float*)d_in[26];
  const float* Wp = (const float*)d_in[27];
  const float* bp = (const float*)d_in[28];
  float* out = (float*)d_out;

  char* ws = (char*)d_ws;
  size_t off = 0;
  auto take = [&](size_t bytes) -> char* {
    char* p = ws + off;
    off = (off + bytes + 255) & ~(size_t)255;
    return p;
  };
  const size_t HSZ = (size_t)MROWS * HD;
  bhalf* wt   = (bhalf*)take((size_t)3145728 * 2);  // per-block bf16 weight slot
  float* hbuf = (float*)take(HSZ * 4);
  bhalf* hbf  = (bhalf*)take(HSZ * 2);
  bhalf* qb   = (bhalf*)take(HSZ * 2);
  bhalf* kb   = (bhalf*)take(HSZ * 2);
  bhalf* vb   = (bhalf*)take(HSZ * 2);
  bhalf* vtb  = (bhalf*)take(HSZ * 2);
  bhalf* ctxb = (bhalf*)take(HSZ * 2);
  float* tmp  = (float*)take(HSZ * 4);
  float* xbuf = (float*)take(HSZ * 4);
  bhalf* mlp  = (bhalf*)take((size_t)MROWS * DFF_ * 2);

  bhalf* wtq = wt;
  bhalf* wtk = wt +  262144;
  bhalf* wtv = wt +  524288;
  bhalf* wto = wt +  786432;
  bhalf* wt1 = wt + 1048576;
  bhalf* wt2 = wt + 2097152;

  // Embedding + input LN
  embed_ln_kernel<<<MROWS, 256, 0, stream>>>(timesteps, state0, state1, actions,
      time_emb, Ws, bs, Wa, ba, elng, elnb, hbuf, hbf);

  const dim3 gQKV(HD/128,  MROWS/128);   // (4, 48)
  const dim3 gFC1(DFF_/128, MROWS/128);  // (16, 48)

  for (int blk = 0; blk < NBLK_; blk++) {
    // convert this block's weights to bf16 [N][K]
    wcvt_t_kernel<<<(512*512)/256, 256, 0, stream>>>(Wq + (size_t)blk*512*512, wtq, 512, 512);
    wcvt_t_kernel<<<(512*512)/256, 256, 0, stream>>>(Wk + (size_t)blk*512*512, wtk, 512, 512);
    wcvt_t_kernel<<<(512*512)/256, 256, 0, stream>>>(Wv + (size_t)blk*512*512, wtv, 512, 512);
    wcvt_t_kernel<<<(512*512)/256, 256, 0, stream>>>(Wo + (size_t)blk*512*512, wto, 512, 512);
    wcvt_t_kernel<<<(512*2048)/256, 256, 0, stream>>>(W1 + (size_t)blk*512*2048, wt1, 512, 2048);
    wcvt_t_kernel<<<(2048*512)/256, 256, 0, stream>>>(W2 + (size_t)blk*2048*512, wt2, 2048, 512);

    // QKV projections (bf16 out)
    gemm_bf16_kernel<0><<<gQKV, 256, 0, stream>>>(hbf, wtq, bq + blk*HD, qb, nullptr, MROWS, HD, HD);
    gemm_bf16_kernel<0><<<gQKV, 256, 0, stream>>>(hbf, wtk, bk + blk*HD, kb, nullptr, MROWS, HD, HD);
    gemm_bf16_kernel<0><<<gQKV, 256, 0, stream>>>(hbf, wtv, bv + blk*HD, vb, nullptr, MROWS, HD, HD);

    // attention
    vtrans_kernel<<<(MROWS*HD)/256, 256, 0, stream>>>(vb, vtb);
    attn_kernel<<<(B_*NH_*(S_/16))/4, 128, 0, stream>>>(qb, kb, vtb, ctxb);

    // output projection + residual, then LN1 -> xbuf (f32) + hbf (bf16)
    gemm_bf16_kernel<2><<<gQKV, 256, 0, stream>>>(ctxb, wto, bo + blk*HD, tmp, hbuf, MROWS, HD, HD);
    ln_kernel<<<MROWS, 256, 0, stream>>>(tmp, ln1g + blk*HD, ln1b + blk*HD, xbuf, hbf);

    // MLP: fc1 + GELU (bf16), fc2 + residual (f32), LN2 -> hbuf + hbf
    gemm_bf16_kernel<1><<<gFC1, 256, 0, stream>>>(hbf, wt1, b1 + blk*DFF_, mlp, nullptr, MROWS, DFF_, HD);
    gemm_bf16_kernel<2><<<gQKV, 256, 0, stream>>>(mlp, wt2, b2 + blk*HD, tmp, xbuf, MROWS, HD, DFF_);
    ln_kernel<<<MROWS, 256, 0, stream>>>(tmp, ln2g + blk*HD, ln2b + blk*HD, hbuf, hbf);
  }

  // final projection of stream-1 rows: 512 -> 32
  fproj_kernel<<<B_*T_, 32, 0, stream>>>(hbuf, Wp, bp, out);
}
